// BastileGptOssExperts_2430951490195
// MI455X (gfx1250) — compile-verified, matrix operand order checked by reading, not run on hardware
//
#include <hip/hip_runtime.h>

// ---------------------------------------------------------------------------
// Dense GPT-OSS MoE experts for MI455X (gfx1250), bf16 WMMA path.
//
// Pass 1: per expert, gate_up GEMM (fp32 inputs -> bf16 in LDS) + GEGLU,
//         pre-scaled by routing weight, written bf16 to workspace [T, E*D].
// Pass 2: single GEMM over K = E*D = 16384 (expert sum folded into K since
//         down_proj is (E,D,H) contiguous), + bias mix epilogue, fp32 out.
//         A-tiles staged with CDNA5 async global->LDS copies (ASYNCcnt).
// ---------------------------------------------------------------------------

typedef __attribute__((ext_vector_type(16))) __bf16 v16bf;
typedef __attribute__((ext_vector_type(8)))  float  v8f;

#define T_TOK 4096            // B*S
#define H_DIM 2048
#define D_DIM 2048
#define E_EXP 8
#define GU_DIM 4096           // 2*D
#define K_TOT  (E_EXP * D_DIM) // 16384

#define BM 128
#define BN 128
#define BK 32
#define LDSK 40               // padded K stride in bf16 elements (80 B, 16B-aligned)

__device__ __forceinline__ unsigned short f2bf(float f) {
    // round-to-nearest-even fp32 -> bf16
    unsigned int u = __float_as_uint(f);
    u += 0x7FFFu + ((u >> 16) & 1u);
    return (unsigned short)(u >> 16);
}

union Frag { v16bf v; uint4 q[2]; };

// ---------------------------------------------------------------------------
// Kernel A: gate_up = x @ W1_e + b1_e ; act = geglu(gate_up) * rw[t,e]
// writes bf16 act to ws laid out [T][E*D] (e-major K for pass 2)
// ---------------------------------------------------------------------------
__global__ __launch_bounds__(256)
void moe_gateup_geglu(const float* __restrict__ x,
                      const float* __restrict__ w1,
                      const float* __restrict__ b1,
                      const float* __restrict__ rw,
                      unsigned short* __restrict__ act)
{
    __shared__ unsigned short As[2][BM][LDSK];
    __shared__ unsigned short Bs[2][BN][LDSK];

    const int tid   = threadIdx.x;
    const int lane  = tid & 31;
    const int wv    = tid >> 5;
    const int waveM = wv >> 2;          // 0..1  (64-row strip)
    const int waveN = wv & 3;           // 0..3  (32-col strip)
    const int l16   = lane & 15;
    const int half  = lane >> 4;
    const int koff  = half * 8;         // bf16 element offset of first K-chunk

    const int e  = blockIdx.z;
    const int m0 = blockIdx.y * BM;     // token base
    const int n0 = blockIdx.x * BN;     // gate_up column base (even)

    const float* w1e = w1 + (size_t)e * H_DIM * GU_DIM;

    auto load_tiles = [&](int k0, int buf) {
        // A tile: x[m0..m0+127][k0..k0+31]  fp32 -> bf16
        #pragma unroll
        for (int j = 0; j < 4; ++j) {
            int idx = tid + j * 256;                 // 0..1023
            int row = idx >> 3;                      // 0..127
            int c4  = (idx & 7) * 4;                 // 0..28
            const float4 v = *(const float4*)(x + (size_t)(m0 + row) * H_DIM + k0 + c4);
            unsigned int lo = (unsigned int)f2bf(v.x) | ((unsigned int)f2bf(v.y) << 16);
            unsigned int hi = (unsigned int)f2bf(v.z) | ((unsigned int)f2bf(v.w) << 16);
            *(uint2*)&As[buf][row][c4] = make_uint2(lo, hi);
        }
        // B tile: W1_e[k0..k0+31][n0..n0+127] fp32 -> bf16, transposed to [n][k]
        #pragma unroll
        for (int j = 0; j < 4; ++j) {
            int idx = tid + j * 256;
            int kk  = idx >> 5;                      // 0..31
            int nn  = (idx & 31) * 4;                // 0..124
            const float* src = w1e + (size_t)(k0 + kk) * GU_DIM + n0 + nn;
            __builtin_prefetch(src + (size_t)BK * GU_DIM, 0, 1); // next K-tile -> global_prefetch_b8
            const float4 v = *(const float4*)src;
            Bs[buf][nn + 0][kk] = f2bf(v.x);
            Bs[buf][nn + 1][kk] = f2bf(v.y);
            Bs[buf][nn + 2][kk] = f2bf(v.z);
            Bs[buf][nn + 3][kk] = f2bf(v.w);
        }
    };

    v8f acc[4][2];
    #pragma unroll
    for (int i = 0; i < 4; ++i)
        #pragma unroll
        for (int j = 0; j < 2; ++j)
            acc[i][j] = (v8f){0.f,0.f,0.f,0.f,0.f,0.f,0.f,0.f};

    load_tiles(0, 0);
    __syncthreads();

    const int nk = H_DIM / BK;          // 64
    for (int kt = 0; kt < nk; ++kt) {
        const int cur = kt & 1;

        // Load ALL fragments first (12x ds_load_b128 issue back-to-back),
        // then run the 8 WMMAs as a dense clause without dscnt stalls.
        Frag bfg[2];
        #pragma unroll
        for (int nsub = 0; nsub < 2; ++nsub) {
            const int brow = waveN * 32 + nsub * 16 + l16;
            bfg[nsub].q[0] = *(const uint4*)&Bs[cur][brow][koff];
            bfg[nsub].q[1] = *(const uint4*)&Bs[cur][brow][koff + 16];
        }
        Frag af[4];
        #pragma unroll
        for (int msub = 0; msub < 4; ++msub) {
            const int arow = waveM * 64 + msub * 16 + l16;
            af[msub].q[0] = *(const uint4*)&As[cur][arow][koff];
            af[msub].q[1] = *(const uint4*)&As[cur][arow][koff + 16];
        }

        if (kt + 1 < nk) load_tiles((kt + 1) * BK, cur ^ 1);

        #pragma unroll
        for (int msub = 0; msub < 4; ++msub)
            #pragma unroll
            for (int nsub = 0; nsub < 2; ++nsub)
                acc[msub][nsub] = __builtin_amdgcn_wmma_f32_16x16x32_bf16(
                    false, af[msub].v, false, bfg[nsub].v, (short)0,
                    acc[msub][nsub], false, false);

        __syncthreads();
    }

    // Epilogue: +bias, GEGLU (even cols = gate, odd = up), * routing weight
    const float ALPHA = 1.702f, LIMIT = 7.0f;
    #pragma unroll
    for (int nsub = 0; nsub < 2; ++nsub) {
        const int n_g  = n0 + waveN * 32 + nsub * 16 + l16;   // gate_up column
        const float bias = b1[e * GU_DIM + n_g];
        #pragma unroll
        for (int msub = 0; msub < 4; ++msub) {
            const int rbase = m0 + waveM * 64 + msub * 16 + half * 8;
            #pragma unroll
            for (int r = 0; r < 8; ++r) {
                const int t = rbase + r;
                float v = acc[msub][nsub][r] + bias;
                float p = __shfl_xor(v, 1, 32);   // partner column (gate<->up)
                if ((lane & 1) == 0) {
                    float g = fminf(v, LIMIT);
                    float u = fminf(fmaxf(p, -LIMIT), LIMIT);
                    float glu = g / (1.0f + __expf(-ALPHA * g));
                    float a = (u + 1.0f) * glu * rw[t * E_EXP + e];
                    act[(size_t)t * K_TOT + e * D_DIM + (n_g >> 1)] = f2bf(a);
                }
            }
        }
    }
}

// ---------------------------------------------------------------------------
// Kernel B: out[t,h] = act[t,:] @ W2flat[:,h] + sum_e rw[t,e]*b2[e,h]
// A-tile staged via CDNA5 async global->LDS (no VGPR round-trip, ASYNCcnt).
// ---------------------------------------------------------------------------
__global__ __launch_bounds__(256)
void moe_down_combine(const unsigned short* __restrict__ act,
                      const float* __restrict__ w2,   // (E*D, H) row-major
                      const float* __restrict__ b2,   // (E, H)
                      const float* __restrict__ rw,   // (T, E)
                      float* __restrict__ out)
{
    __shared__ unsigned short As[2][BM][LDSK];
    __shared__ unsigned short Bs[2][BN][LDSK];

    const int tid   = threadIdx.x;
    const int lane  = tid & 31;
    const int wv    = tid >> 5;
    const int waveM = wv >> 2;
    const int waveN = wv & 3;
    const int l16   = lane & 15;
    const int half  = lane >> 4;
    const int koff  = half * 8;

    const int m0 = blockIdx.y * BM;     // token base
    const int n0 = blockIdx.x * BN;     // hidden (H) column base

    // A tile: act bf16, direct copy -> async global->LDS b128
    auto load_a_async = [&](int k0, int buf) {
        #pragma unroll
        for (int j = 0; j < 2; ++j) {
            int idx = tid + j * 256;                 // 0..511
            int row = idx >> 2;                      // 0..127
            int c8  = (idx & 3) * 8;                 // 0..24 (bf16 elems)
            unsigned lds_off =
                (unsigned)(size_t)&As[buf][row][c8]; // low 32b of LDS aperture addr
            const unsigned short* g =
                act + (size_t)(m0 + row) * K_TOT + k0 + c8;
            asm volatile("global_load_async_to_lds_b128 %0, %1, off"
                         :: "v"(lds_off), "v"((unsigned long long)(size_t)g)
                         : "memory");
        }
    };
    // B tile: W2flat[k0..k0+31][n0..n0+127] fp32 -> bf16, transposed [n][k]
    auto load_b = [&](int k0, int buf) {
        #pragma unroll
        for (int j = 0; j < 4; ++j) {
            int idx = tid + j * 256;
            int kk  = idx >> 5;
            int nn  = (idx & 31) * 4;
            const float* src = w2 + (size_t)(k0 + kk) * H_DIM + n0 + nn;
            __builtin_prefetch(src + (size_t)BK * H_DIM, 0, 1);
            const float4 v = *(const float4*)src;
            Bs[buf][nn + 0][kk] = f2bf(v.x);
            Bs[buf][nn + 1][kk] = f2bf(v.y);
            Bs[buf][nn + 2][kk] = f2bf(v.z);
            Bs[buf][nn + 3][kk] = f2bf(v.w);
        }
    };

    v8f acc[4][2];
    #pragma unroll
    for (int i = 0; i < 4; ++i)
        #pragma unroll
        for (int j = 0; j < 2; ++j)
            acc[i][j] = (v8f){0.f,0.f,0.f,0.f,0.f,0.f,0.f,0.f};

    load_a_async(0, 0);
    load_b(0, 0);
    asm volatile("s_wait_asynccnt 0x0" ::: "memory");
    __syncthreads();

    const int nk = K_TOT / BK;          // 512
    for (int kt = 0; kt < nk; ++kt) {
        const int cur = kt & 1;

        // fire async copies for the next A tile first (pure DMA, no VGPRs)
        if (kt + 1 < nk) load_a_async((kt + 1) * BK, cur ^ 1);

        Frag bfg[2];
        #pragma unroll
        for (int nsub = 0; nsub < 2; ++nsub) {
            const int brow = waveN * 32 + nsub * 16 + l16;
            bfg[nsub].q[0] = *(const uint4*)&Bs[cur][brow][koff];
            bfg[nsub].q[1] = *(const uint4*)&Bs[cur][brow][koff + 16];
        }
        Frag af[4];
        #pragma unroll
        for (int msub = 0; msub < 4; ++msub) {
            const int arow = waveM * 64 + msub * 16 + l16;
            af[msub].q[0] = *(const uint4*)&As[cur][arow][koff];
            af[msub].q[1] = *(const uint4*)&As[cur][arow][koff + 16];
        }

        if (kt + 1 < nk) load_b((kt + 1) * BK, cur ^ 1);

        #pragma unroll
        for (int msub = 0; msub < 4; ++msub)
            #pragma unroll
            for (int nsub = 0; nsub < 2; ++nsub)
                acc[msub][nsub] = __builtin_amdgcn_wmma_f32_16x16x32_bf16(
                    false, af[msub].v, false, bfg[nsub].v, (short)0,
                    acc[msub][nsub], false, false);

        asm volatile("s_wait_asynccnt 0x0" ::: "memory");
        __syncthreads();
    }

    // Epilogue: bias mix sum_e rw[t,e] * b2[e,h], fp32 store
    #pragma unroll
    for (int nsub = 0; nsub < 2; ++nsub) {
        const int h = n0 + waveN * 32 + nsub * 16 + l16;
        float cb[E_EXP];
        #pragma unroll
        for (int ee = 0; ee < E_EXP; ++ee) cb[ee] = b2[ee * H_DIM + h];
        #pragma unroll
        for (int msub = 0; msub < 4; ++msub) {
            const int rbase = m0 + waveM * 64 + msub * 16 + half * 8;
            #pragma unroll
            for (int r = 0; r < 8; ++r) {
                const int t = rbase + r;
                float bias = 0.f;
                #pragma unroll
                for (int ee = 0; ee < E_EXP; ++ee)
                    bias += rw[t * E_EXP + ee] * cb[ee];
                out[(size_t)t * H_DIM + h] = acc[msub][nsub][r] + bias;
            }
        }
    }
}

// ---------------------------------------------------------------------------
extern "C" void kernel_launch(void* const* d_in, const int* in_sizes, int n_in,
                              void* d_out, int out_size, void* d_ws, size_t ws_size,
                              hipStream_t stream) {
    const float* x   = (const float*)d_in[0];   // (B,S,H) = (T,H)
    const float* w1  = (const float*)d_in[1];   // (E,H,2D)
    const float* b1  = (const float*)d_in[2];   // (E,2D)
    const float* w2  = (const float*)d_in[3];   // (E,D,H) == (E*D, H)
    const float* b2  = (const float*)d_in[4];   // (E,H)
    const float* rw  = (const float*)d_in[5];   // (T,E)
    // d_in[6] = router_indices: unused (reference is dense over all experts)
    float* out = (float*)d_out;

    const size_t need = (size_t)T_TOK * K_TOT * sizeof(unsigned short); // 128 MiB
    if (ws_size < need) return;
    unsigned short* actbuf = (unsigned short*)d_ws;

    dim3 grdA(GU_DIM / BN, T_TOK / BM, E_EXP);   // 32 x 32 x 8
    moe_gateup_geglu<<<grdA, dim3(256), 0, stream>>>(x, w1, b1, rw, actbuf);

    dim3 grdB(H_DIM / BN, T_TOK / BM, 1);        // 16 x 32
    moe_down_combine<<<grdB, dim3(256), 0, stream>>>(actbuf, w2, b2, rw, out);
}
